// APT_ATTN_Block_51694226375277
// MI455X (gfx1250) — matrix operand order repaired and checked by reading, not compiled, not run on hardware
//
#include <hip/hip_runtime.h>

typedef __bf16 bf16;
typedef __attribute__((ext_vector_type(16))) __bf16 v16bf;
typedef __attribute__((ext_vector_type(8)))  __bf16 v8bf;
typedef __attribute__((ext_vector_type(8)))  float  v8f;

#define DIMD 1024
#define DIMP 128
#define NB   4096
#define NKEYS 8192            // NS == NT == 8192 (compile-time so V-tile offsets fold to immediates)
#define NCLS 1000
#define NROWS (NB + 2 * NKEYS)   // 20480

// ---------- bf16 helpers (round-to-nearest-even) ----------
__device__ __forceinline__ bf16 f2bf(float f) {
  union { float f; unsigned u; } c; c.f = f;
  unsigned r = (c.u + 0x7FFFu + ((c.u >> 16) & 1u)) >> 16;
  union { unsigned short s; bf16 b; } o; o.s = (unsigned short)r;
  return o.b;
}
__device__ __forceinline__ float bf2f(bf16 b) {
  union { bf16 b; unsigned short s; } i; i.b = b;
  union { unsigned u; float f; } o; o.u = ((unsigned)i.s) << 16;
  return o.f;
}
__device__ __forceinline__ v16bf cat8(v8bf lo, v8bf hi) {
  return __builtin_shufflevector(lo, hi, 0,1,2,3,4,5,6,7,8,9,10,11,12,13,14,15);
}
#define WMMA_BF16(a, b, c) \
  __builtin_amdgcn_wmma_f32_16x16x32_bf16(false, (a), false, (b), (short)0, (c), false, false)

// ---------- cast kernels ----------
__global__ void cast_f32_bf16(const float* __restrict__ s, bf16* __restrict__ d, int n) {
  int i = blockIdx.x * 256 + threadIdx.x;
  if (i < n) d[i] = f2bf(s[i]);
}
// W3 is (3*D, P); q uses rows 3d, k rows 3d+1, v rows 3d+2
__global__ void split_w3(const float* __restrict__ W3, bf16* __restrict__ Wq,
                         bf16* __restrict__ Wk, bf16* __restrict__ Wv) {
  int i = blockIdx.x * 256 + threadIdx.x;       // over D*P = 131072
  int d = i >> 7, p = i & 127;
  size_t b = (size_t)(3 * d) * DIMP + p;
  Wq[i] = f2bf(W3[b]);
  Wk[i] = f2bf(W3[b + DIMP]);
  Wv[i] = f2bf(W3[b + 2 * DIMP]);
}
// Ft (1000 x 1024) -> bf16 padded to (1024 x 1024), pad rows = 0
__global__ void cast_ft_pad(const float* __restrict__ Ft, bf16* __restrict__ d) {
  int i = blockIdx.x * 256 + threadIdx.x;       // over 1024*1024
  int r = i >> 10;
  d[i] = (r < NCLS) ? f2bf(Ft[i]) : f2bf(0.f);
}

// ---------- generic WMMA GEMM: out = epi( A(MxK) @ Bt(NxK)^T ) ----------
// Block covers 64(M) x 128(N): 8 waves as 2x4 grid, 32x32 per wave.
// EPI: 0 none, 1 +bias(p0, strided), 2 +bias then BN(eval)+ReLU, 3 *exp(*sc)
// OUT: 0 f32 row-major, 1 bf16 row-major, 2 bf16 transposed (out[n*ldo + m], vector store)
template<int EPI, int OUT, int LDA, int LDB>
__global__ void gemm_k(const bf16* __restrict__ A, const bf16* __restrict__ Bt,
                       int Npad, int Nreal, int K,
                       void* __restrict__ out, int ldo,
                       const float* __restrict__ p0, int p0s, int p0o,
                       const float* __restrict__ pg, const float* __restrict__ pb,
                       const float* __restrict__ pm, const float* __restrict__ pv,
                       const float* __restrict__ sc) {
  int lane = threadIdx.x & 31, wave = threadIdx.x >> 5;
  int m0 = blockIdx.x * 64 + (wave >> 2) * 32;
  int n0 = blockIdx.y * 128 + (wave & 3) * 32;
  if (n0 >= Npad) return;                        // wave-uniform exit, EXEC stays all-ones

  int r16 = lane & 15;
  int ako = (lane >> 4) * 8;                     // A-operand K sub-offset
  int bko = (lane >> 4) * 8;                     // B-operand K sub-offset

  // streaming per-lane pointers; all tile deltas are compile-time immediates
  const bf16* ptrA = A + (size_t)(m0 + r16) * LDA + ako;
  const bf16* ptrB = Bt + (size_t)(n0 + r16) * LDB + bko;

  v8f c[2][2];
  for (int a = 0; a < 2; a++) for (int b = 0; b < 2; b++)
    for (int i = 0; i < 8; i++) c[a][b][i] = 0.f;

  for (int k = 0; k < K; k += 32) {
    v16bf a0 = cat8(*(const v8bf*)(ptrA),            *(const v8bf*)(ptrA + 16));
    v16bf a1 = cat8(*(const v8bf*)(ptrA + 16 * LDA), *(const v8bf*)(ptrA + 16 * LDA + 16));
    v16bf b0 = cat8(*(const v8bf*)(ptrB),            *(const v8bf*)(ptrB + 16));
    v16bf b1 = cat8(*(const v8bf*)(ptrB + 16 * LDB), *(const v8bf*)(ptrB + 16 * LDB + 16));
    c[0][0] = WMMA_BF16(a0, b0, c[0][0]);
    c[0][1] = WMMA_BF16(a0, b1, c[0][1]);
    c[1][0] = WMMA_BF16(a1, b0, c[1][0]);
    c[1][1] = WMMA_BF16(a1, b1, c[1][1]);
    ptrA += 32; ptrB += 32;
  }

  int n16 = lane & 15, mb = (lane >> 4) * 8;
  float scale = 1.f;
  if (EPI == 3) scale = __expf(sc[0]);
  for (int tm = 0; tm < 2; tm++) {
    for (int tn = 0; tn < 2; tn++) {
      int ng = n0 + tn * 16 + n16;
      float bbias = 0.f, gg = 1.f, beb = 0.f, mm = 0.f, rs = 1.f;
      if (EPI == 1) bbias = p0[(size_t)ng * p0s + p0o];
      if (EPI == 2) {
        bbias = p0[(size_t)ng * p0s + p0o];
        gg = pg[ng]; beb = pb[ng]; mm = pm[ng];
        rs = rsqrtf(pv[ng] + 1e-5f);
      }
      if (OUT == 2) {
        // lane's 8 accum values are 8 consecutive m at column ng -> one b128 store
        v8bf pk;
        for (int r = 0; r < 8; r++) {
          float y = c[tm][tn][r];
          if (EPI == 1) y += bbias;
          pk[r] = f2bf(y);
        }
        *(v8bf*)((bf16*)out + (size_t)ng * ldo + (m0 + tm * 16 + mb)) = pk;
      } else {
        for (int r = 0; r < 8; r++) {
          int m = m0 + tm * 16 + mb + r;
          float y = c[tm][tn][r];
          if (EPI == 1) y += bbias;
          if (EPI == 2) { y = (y + bbias - mm) * rs * gg + beb; y = y > 0.f ? y : 0.f; }
          if (EPI == 3) y *= scale;
          if (OUT == 0) { if (ng < Nreal) ((float*)out)[(size_t)m * ldo + ng] = y; }
          else          { ((bf16*)out)[(size_t)m * ldo + ng] = f2bf(y); }
        }
      }
    }
  }
}

// ---------- flash attention: O = softmax(0.1 * Q K^T) V, streamed over 8192 keys ----------
// Q: (NB x D) bf16; Kb: (NKEYS x D) bf16; VT: (D x NKEYS) bf16.
// 8 waves/block; wave w owns output features [w*128, w*128+128).
// QK^T split across waves over D; partial S reduced via LDS ds_add_f32 atomics.
__global__ void flash_attn(const bf16* __restrict__ Qb, const bf16* __restrict__ Kb,
                           const bf16* __restrict__ VT, bf16* __restrict__ Obf) {
  __shared__ float sS[16 * 32];
  __shared__ float sM[16], sL[16], sA[16];
  int tid = threadIdx.x;
  int lane = tid & 31, wave = tid >> 5;
  int f0 = wave * 128;
  int qrow0 = blockIdx.x * 16;
  int n16 = lane & 15, mb = (lane >> 4) * 8;
  int ako = (lane >> 4) * 8;
  int bko = (lane >> 4) * 8;

  // resident Q A-operand tiles (16 rows x 128 features per wave)
  v16bf qa[4];
  {
    const bf16* qp = Qb + (size_t)(qrow0 + n16) * DIMD + f0 + ako;
#pragma unroll
    for (int kk = 0; kk < 4; kk++)
      qa[kk] = cat8(*(const v8bf*)(qp + kk * 32), *(const v8bf*)(qp + kk * 32 + 16));
  }
  v8f acc[8];
  for (int t = 0; t < 8; t++) for (int r = 0; r < 8; r++) acc[t][r] = 0.f;
  if (tid < 16) { sM[tid] = -3.0e38f; sL[tid] = 0.f; }

  // streaming per-lane pointers; all tile deltas are immediates (NKEYS/DIMD compile-time)
  const bf16* kp = Kb + (size_t)n16 * DIMD + f0 + bko;          // += 32*DIMD per block
  const bf16* vp = VT + (size_t)(f0 + n16) * NKEYS + bko;       // += 32 per block

  for (int kb = 0; kb < NKEYS; kb += 32) {
    __syncthreads();
    sS[tid] = 0.f; sS[tid + 256] = 0.f;
    __syncthreads();

    // partial S = Q[:, f0:f0+128] @ K[kb:kb+32, f0:f0+128]^T  (batched tile loads)
    v16bf kt[8];
#pragma unroll
    for (int kk = 0; kk < 4; kk++) {
      kt[kk]     = cat8(*(const v8bf*)(kp + kk * 32), *(const v8bf*)(kp + kk * 32 + 16));
      kt[4 + kk] = cat8(*(const v8bf*)(kp + 16 * DIMD + kk * 32), *(const v8bf*)(kp + 16 * DIMD + kk * 32 + 16));
    }
    v8f s0, s1;
    for (int i = 0; i < 8; i++) { s0[i] = 0.f; s1[i] = 0.f; }
#pragma unroll
    for (int kk = 0; kk < 4; kk++) {
      s0 = WMMA_BF16(qa[kk], kt[kk], s0);
      s1 = WMMA_BF16(qa[kk], kt[4 + kk], s1);
    }
#pragma unroll
    for (int r = 0; r < 8; r++) {
      atomicAdd(&sS[(mb + r) * 32 + n16], s0[r]);
      atomicAdd(&sS[(mb + r) * 32 + 16 + n16], s1[r]);
    }
    __syncthreads();

    // online softmax update per query row
    if (tid < 16) {
      float mo = sM[tid], mx = mo;
      for (int j = 0; j < 32; j++) { float v = 0.1f * sS[tid * 32 + j]; if (v > mx) mx = v; }
      float al = __expf(mo - mx);
      float sum = 0.f;
      for (int j = 0; j < 32; j++) {
        float p = __expf(0.1f * sS[tid * 32 + j] - mx);
        sS[tid * 32 + j] = p; sum += p;
      }
      sL[tid] = sL[tid] * al + sum; sM[tid] = mx; sA[tid] = al;
    }
    __syncthreads();

    float alph[8];
#pragma unroll
    for (int r = 0; r < 8; r++) alph[r] = sA[mb + r];
    // repack P (16x32) from LDS into WMMA A layout
    v16bf pmat;
    {
      int pr = lane & 15, pk2 = (lane >> 4) * 8;
#pragma unroll
      for (int i = 0; i < 8; i++) {
        pmat[i]     = f2bf(sS[pr * 32 + pk2 + i]);
        pmat[8 + i] = f2bf(sS[pr * 32 + pk2 + 16 + i]);
      }
    }
    // batched V tile loads (immediate offsets t*16*NKEYS), then rescale+accumulate
    v16bf bv[8];
#pragma unroll
    for (int t = 0; t < 8; t++) bv[t] = cat8(*(const v8bf*)(vp + t * 16 * NKEYS), *(const v8bf*)(vp + t * 16 * NKEYS + 16));
#pragma unroll
    for (int t = 0; t < 8; t++) {
      for (int r = 0; r < 8; r++) acc[t][r] *= alph[r];
      acc[t] = WMMA_BF16(pmat, bv[t], acc[t]);
    }
    kp += 32 * DIMD;
    vp += 32;
  }
  __syncthreads();
  float linv[8];
  for (int r = 0; r < 8; r++) linv[r] = 1.f / sL[mb + r];
  for (int t = 0; t < 8; t++)
    for (int r = 0; r < 8; r++) {
      int m = qrow0 + mb + r;
      int cc = f0 + t * 16 + n16;
      Obf[(size_t)m * DIMD + cc] = f2bf(acc[t][r] * linv[r]);
    }
}

// ---------- Fsa/Fta assembly: add Fv, L2-normalize each, sum -> H ----------
__global__ void combine_norm(const float* __restrict__ Fv, const bf16* __restrict__ Gs,
                             const bf16* __restrict__ Gt, bf16* __restrict__ H) {
  __shared__ float r1[256], r2[256];
  int row = blockIdx.x, t = threadIdx.x;
  float fs[4], ft[4], ss = 0.f, st = 0.f;
  for (int j = 0; j < 4; j++) {
    int i = t + j * 256;
    float a = Fv[(size_t)row * DIMD + i];
    float x = a + bf2f(Gs[(size_t)row * DIMD + i]);
    float y = a + bf2f(Gt[(size_t)row * DIMD + i]);
    fs[j] = x; ft[j] = y; ss += x * x; st += y * y;
  }
  r1[t] = ss; r2[t] = st;
  __syncthreads();
  for (int s = 128; s > 0; s >>= 1) {
    if (t < s) { r1[t] += r1[t + s]; r2[t] += r2[t + s]; }
    __syncthreads();
  }
  float inv_s = rsqrtf(r1[0] + 1e-30f);
  float inv_t = rsqrtf(r2[0] + 1e-30f);
  for (int j = 0; j < 4; j++) {
    int i = t + j * 256;
    H[(size_t)row * DIMD + i] = f2bf(fs[j] * inv_s + ft[j] * inv_t);
  }
}

// ---------------------------------------------------------------------------
extern "C" void kernel_launch(void* const* d_in, const int* in_sizes, int n_in,
                              void* d_out, int out_size, void* d_ws, size_t ws_size,
                              hipStream_t stream) {
  const float* Ft  = (const float*)d_in[0];
  const float* Fv  = (const float*)d_in[1];
  const float* Fvs = (const float*)d_in[2];
  const float* Fvt = (const float*)d_in[3];
  const float* W1  = (const float*)d_in[4];
  const float* b1  = (const float*)d_in[5];
  const float* g1f = (const float*)d_in[6];
  const float* be1 = (const float*)d_in[7];
  const float* m1  = (const float*)d_in[8];
  const float* v1  = (const float*)d_in[9];
  const float* W2  = (const float*)d_in[10];
  const float* b2  = (const float*)d_in[11];
  const float* g2f = (const float*)d_in[12];
  const float* be2 = (const float*)d_in[13];
  const float* m2  = (const float*)d_in[14];
  const float* v2  = (const float*)d_in[15];
  const float* W3  = (const float*)d_in[16];
  const float* b3  = (const float*)d_in[17];
  const float* Wp  = (const float*)d_in[18];
  const float* bp  = (const float*)d_in[19];
  const float* logit_scale = (const float*)d_in[20];

  // workspace bump allocator (256B aligned)
  char* base = (char*)d_ws;
  size_t off = 0;
  auto alloc = [&](size_t bytes) -> void* {
    void* p = base + off;
    off = (off + bytes + 255) & ~(size_t)255;
    return p;
  };
  bf16* Xbf  = (bf16*)alloc((size_t)NROWS * DIMD * 2);
  bf16* W1b  = (bf16*)alloc((size_t)DIMP * DIMD * 2);
  bf16* W2b  = (bf16*)alloc((size_t)DIMP * DIMP * 2);
  bf16* Wqb  = (bf16*)alloc((size_t)DIMD * DIMP * 2);
  bf16* Wkb  = (bf16*)alloc((size_t)DIMD * DIMP * 2);
  bf16* Wvb  = (bf16*)alloc((size_t)DIMD * DIMP * 2);
  bf16* Wpb  = (bf16*)alloc((size_t)DIMD * DIMD * 2);
  bf16* Ftb  = (bf16*)alloc((size_t)1024 * DIMD * 2);
  bf16* h1   = (bf16*)alloc((size_t)NROWS * DIMP * 2);
  bf16* h2   = (bf16*)alloc((size_t)NROWS * DIMP * 2);
  bf16* qb   = (bf16*)alloc((size_t)NB * DIMD * 2);
  bf16* Ksb  = (bf16*)alloc((size_t)NKEYS * DIMD * 2);
  bf16* VsTb = (bf16*)alloc((size_t)DIMD * NKEYS * 2);
  bf16* Ktb  = (bf16*)alloc((size_t)NKEYS * DIMD * 2);
  bf16* VtTb = (bf16*)alloc((size_t)DIMD * NKEYS * 2);
  bf16* Osb  = (bf16*)alloc((size_t)NB * DIMD * 2);
  bf16* Otb  = (bf16*)alloc((size_t)NB * DIMD * 2);
  bf16* Gsb  = (bf16*)alloc((size_t)NB * DIMD * 2);
  bf16* Gtb  = (bf16*)alloc((size_t)NB * DIMD * 2);
  bf16* Hb   = (bf16*)alloc((size_t)NB * DIMD * 2);

  // --- casts ---
  cast_f32_bf16<<<(NB    * DIMD) / 256, 256, 0, stream>>>(Fv,  Xbf,                          NB    * DIMD);
  cast_f32_bf16<<<(NKEYS * DIMD) / 256, 256, 0, stream>>>(Fvs, Xbf + (size_t)NB * DIMD,      NKEYS * DIMD);
  cast_f32_bf16<<<(NKEYS * DIMD) / 256, 256, 0, stream>>>(Fvt, Xbf + (size_t)(NB + NKEYS) * DIMD, NKEYS * DIMD);
  cast_f32_bf16<<<(DIMP * DIMD) / 256, 256, 0, stream>>>(W1, W1b, DIMP * DIMD);
  cast_f32_bf16<<<(DIMP * DIMP) / 256, 256, 0, stream>>>(W2, W2b, DIMP * DIMP);
  cast_f32_bf16<<<(DIMD * DIMD) / 256, 256, 0, stream>>>(Wp, Wpb, DIMD * DIMD);
  split_w3<<<(DIMD * DIMP) / 256, 256, 0, stream>>>(W3, Wqb, Wkb, Wvb);
  cast_ft_pad<<<(1024 * DIMD) / 256, 256, 0, stream>>>(Ft, Ftb);

  // --- pre_project MLP (all 20480 rows) ---
  gemm_k<2, 1, DIMD, DIMD><<<dim3(NROWS / 64, 1), 256, 0, stream>>>(
      Xbf, W1b, DIMP, DIMP, DIMD, h1, DIMP, b1, 1, 0, g1f, be1, m1, v1, nullptr);
  gemm_k<2, 1, DIMP, DIMP><<<dim3(NROWS / 64, 1), 256, 0, stream>>>(
      h1, W2b, DIMP, DIMP, DIMP, h2, DIMP, b2, 1, 0, g2f, be2, m2, v2, nullptr);

  // --- q / k / v heads (bias from strided b3) ---
  gemm_k<1, 1, DIMP, DIMP><<<dim3(NB / 64, 8), 256, 0, stream>>>(
      h2, Wqb, DIMD, DIMD, DIMP, qb, DIMD, b3, 3, 0, nullptr, nullptr, nullptr, nullptr, nullptr);
  const bf16* h2s = h2 + (size_t)NB * DIMP;
  const bf16* h2t = h2 + (size_t)(NB + NKEYS) * DIMP;
  gemm_k<1, 1, DIMP, DIMP><<<dim3(NKEYS / 64, 8), 256, 0, stream>>>(
      h2s, Wkb, DIMD, DIMD, DIMP, Ksb, DIMD, b3, 3, 1, nullptr, nullptr, nullptr, nullptr, nullptr);
  gemm_k<1, 2, DIMP, DIMP><<<dim3(NKEYS / 64, 8), 256, 0, stream>>>(
      h2s, Wvb, DIMD, DIMD, DIMP, VsTb, NKEYS, b3, 3, 2, nullptr, nullptr, nullptr, nullptr, nullptr);
  gemm_k<1, 1, DIMP, DIMP><<<dim3(NKEYS / 64, 8), 256, 0, stream>>>(
      h2t, Wkb, DIMD, DIMD, DIMP, Ktb, DIMD, b3, 3, 1, nullptr, nullptr, nullptr, nullptr, nullptr);
  gemm_k<1, 2, DIMP, DIMP><<<dim3(NKEYS / 64, 8), 256, 0, stream>>>(
      h2t, Wvb, DIMD, DIMD, DIMP, VtTb, NKEYS, b3, 3, 2, nullptr, nullptr, nullptr, nullptr, nullptr);

  // --- flash attention over both banks ---
  flash_attn<<<NB / 16, 256, 0, stream>>>(qb, Ksb, VsTb, Osb);
  flash_attn<<<NB / 16, 256, 0, stream>>>(qb, Ktb, VtTb, Otb);

  // --- post_project: G = O @ Wp^T + bp ---
  gemm_k<1, 1, DIMD, DIMD><<<dim3(NB / 64, 8), 256, 0, stream>>>(
      Osb, Wpb, DIMD, DIMD, DIMD, Gsb, DIMD, bp, 1, 0, nullptr, nullptr, nullptr, nullptr, nullptr);
  gemm_k<1, 1, DIMD, DIMD><<<dim3(NB / 64, 8), 256, 0, stream>>>(
      Otb, Wpb, DIMD, DIMD, DIMD, Gtb, DIMD, bp, 1, 0, nullptr, nullptr, nullptr, nullptr, nullptr);

  // --- Fsa/Fta normalize + sum ---
  combine_norm<<<NB, 256, 0, stream>>>(Fv, Gsb, Gtb, Hb);

  // --- logits = exp(logit_scale) * H @ Ft^T (padded N=1024, stores gated at 1000) ---
  gemm_k<3, 0, DIMD, DIMD><<<dim3(NB / 64, 8), 256, 0, stream>>>(
      Hb, Ftb, 1024, NCLS, DIMD, d_out, NCLS, nullptr, 0, 0, nullptr, nullptr, nullptr, nullptr, logit_scale);
}
